// RegionAwareContrastiveLearning_70300024701532
// MI455X (gfx1250) — matrix operand order repaired and checked by reading, not compiled, not run on hardware
//
#include <hip/hip_runtime.h>
#include <math.h>

typedef __attribute__((ext_vector_type(16))) _Float16 v16h;
typedef __attribute__((ext_vector_type(8)))  _Float16 v8h;
typedef __attribute__((ext_vector_type(4)))  _Float16 v4h;
typedef __attribute__((ext_vector_type(8)))  float    v8f;

static constexpr int B_ = 2, C_ = 128, D_ = 64, H_ = 64, W_ = 64;
static constexpr int CSTRIDE = D_ * H_ * W_;   // 262144
static constexpr int NP = 64;     // patches per batch (4^3), also "mixed" N
static constexpr int S_ = 100;    // sampled voxels
#define TEMP_INV 10.0f
#define EDGE_THR 0.25f

static __device__ __forceinline__ v16h cat8(v8h lo, v8h hi) {
  return __builtin_shufflevector(lo, hi, 0, 1, 2, 3, 4, 5, 6, 7,
                                 8, 9, 10, 11, 12, 13, 14, 15);
}

static __device__ __forceinline__ v8h pack8(float4 a, float4 b) {
  v8h r;
  r[0] = (_Float16)a.x; r[1] = (_Float16)a.y;
  r[2] = (_Float16)a.z; r[3] = (_Float16)a.w;
  r[4] = (_Float16)b.x; r[5] = (_Float16)b.y;
  r[6] = (_Float16)b.z; r[7] = (_Float16)b.w;
  return r;
}

// ---------------------------------------------------------------------------
// k0: zero the per-patch edge counters (B*NP = 128 ints)
// ---------------------------------------------------------------------------
__global__ void k0_zero(int* __restrict__ edge_counts) {
  edge_counts[threadIdx.x] = 0;
}

// ---------------------------------------------------------------------------
// k1: fused Conv3d(128->32,3x3x3,p=1) + ReLU + Conv3d(32->1,1x1x1) + sigmoid
//     + (prob > 0.25) + per-spatial-patch edge counts.
// Implicit GEMM, one workgroup per (b,z,y) row of 64 x-voxels.
// Stage a 66-wide zero-padded x-row per 32-channel block once per (dz,dy);
// the 3 dx taps are LDS-offset shifts of it. B tiles staged pre-swizzled into
// wave32 WMMA fragment layout. All LDS staging via packed b64/b128 stores.
// 8 waves: Mtile = wave&3 (16 voxels), Ntile = wave>>2 (16 out channels).
// ---------------------------------------------------------------------------
__global__ __launch_bounds__(256) void k1_region(
    const float* __restrict__ in,
    const float* __restrict__ cw1, const float* __restrict__ cb1,
    const float* __restrict__ cw2, const float* __restrict__ cb2,
    int* __restrict__ edge_counts) {
  __shared__ alignas(32) _Float16 A_lds[66][32];        // [x+1][k] (x = -1..64)
  __shared__ alignas(32) _Float16 B_sw[3][2][32][16];   // [dx][Nt][lane][e]
  __shared__ float row_acc[64];
  __shared__ int cnt[4];

  const int tid = threadIdx.x;
  const int lane = tid & 31, wave = tid >> 5;
  const int Mt = wave & 3, Nt = wave >> 2;
  const int half = lane >> 4, mr = lane & 15;

  const int row = blockIdx.x;          // b*4096 + z*64 + y
  const int b = row >> 12;
  const int rem = row & 4095;
  const int z = rem >> 6, y = rem & 63;

  if (tid < 64) row_acc[tid] = 0.f;
  if (tid < 4)  cnt[tid] = 0;

  v8f acc = {0.f, 0.f, 0.f, 0.f, 0.f, 0.f, 0.f, 0.f};

  for (int dzi = 0; dzi < 3; ++dzi) {
    const int zin = z + dzi - 1;
    if ((unsigned)zin >= 64u) continue;                 // uniform per block
    for (int dyi = 0; dyi < 3; ++dyi) {
      const int yin = y + dyi - 1;
      if ((unsigned)yin >= 64u) continue;               // uniform per block
      const int zy3 = (dzi * 3 + dyi) * 3;              // tap base (dx = 0..2)
      for (int cb = 0; cb < 4; ++cb) {
        const int c0 = cb * 32;
        __syncthreads();
        // zero halo columns (x = -1 and x = 64): 8 packed stores
        if (tid < 8) {
          v8h z8 = {};
          *(v8h*)&A_lds[(tid >> 2) ? 65 : 0][(tid & 3) * 8] = z8;
        }
        // stage interior: thread -> x = tid&63, 8 consecutive channels;
        // per-q global loads are wave-coalesced (lanes vary x)
        {
          const int x = tid & 63, kc0 = (tid >> 6) * 8;
          const float* src =
              in + (((b * C_ + (c0 + kc0)) * D_ + zin) * H_ + yin) * W_ + x;
          float tmp[8];
          #pragma unroll
          for (int q = 0; q < 8; ++q) tmp[q] = src[q * CSTRIDE];
          v8h pk;
          #pragma unroll
          for (int q = 0; q < 8; ++q) pk[q] = (_Float16)tmp[q];
          *(v8h*)&A_lds[x + 1][kc0] = pk;
        }
        // stage B pre-swizzled: thread -> (o, 4 consecutive k); the 3 dx
        // weights per (o,c) are contiguous in cw1 (b96 loads)
        {
          const int o = tid >> 3, kc0 = (tid & 7) * 4;
          const int ln = (kc0 >> 4) * 16 + (o & 15);
          const int e0 = kc0 & 15;
          const int nt = o >> 4;
          v4h q0, q1, q2;
          #pragma unroll
          for (int j = 0; j < 4; ++j) {
            const float* wsrc = cw1 + (o * C_ + (c0 + kc0 + j)) * 27 + zy3;
            q0[j] = (_Float16)wsrc[0];
            q1[j] = (_Float16)wsrc[1];
            q2[j] = (_Float16)wsrc[2];
          }
          *(v4h*)&B_sw[0][nt][ln][e0] = q0;
          *(v4h*)&B_sw[1][nt][ln][e0] = q1;
          *(v4h*)&B_sw[2][nt][ln][e0] = q2;
        }
        __syncthreads();
        #pragma unroll
        for (int dxi = 0; dxi < 3; ++dxi) {
          const int x = Mt * 16 + mr + dxi;   // (voxel + dx + 1)
          v8h lo = *(const v8h*)&A_lds[x][half * 16];
          v8h hi = *(const v8h*)&A_lds[x][half * 16 + 8];
          v16h af = cat8(lo, hi);
          v16h bf = *(const v16h*)&B_sw[dxi][Nt][lane][0];
          acc = __builtin_amdgcn_wmma_f32_16x16x32_f16(
              false, af, false, bf, (short)0, acc, false, false);
        }
      }
    }
  }
  __syncthreads();
  // fuse pointwise conv2 partial: relu(acc + b1[o]) * w2[o], reduced over o
  {
    const int o = Nt * 16 + mr;
    const float b1v = cb1[o], w2v = cw2[o];
    #pragma unroll
    for (int i = 0; i < 8; ++i) {
      int x = Mt * 16 + half * 8 + i;
      float h = fmaxf(acc[i] + b1v, 0.f);
      atomicAdd(&row_acc[x], h * w2v);
    }
  }
  __syncthreads();
  if (tid < 64) {
    float v = row_acc[tid] + cb2[0];
    float prob = 1.f / (1.f + expf(-v));
    if (prob > EDGE_THR) atomicAdd(&cnt[tid >> 4], 1);
  }
  __syncthreads();
  if (tid < 4) {
    int n = (z >> 4) * 16 + (y >> 4) * 4 + tid;   // spatial patch index
    atomicAdd(&edge_counts[b * NP + n], cnt[tid]);
  }
}

// ---------------------------------------------------------------------------
// k2: per-(b, channel, spatial-patch) means -> PM[b][c][pidx], both tensors.
// grid = B*C*64 = 16384 blocks of 256 threads. float4 row loads.
// ---------------------------------------------------------------------------
__global__ __launch_bounds__(256) void k2_means(
    const float* __restrict__ a, const float* __restrict__ p,
    float* __restrict__ pma, float* __restrict__ pmp) {
  __shared__ float sA[256], sB[256];
  const int blk = blockIdx.x;           // ((b*128)+c)*64 + pidx
  const int pidx = blk & 63;
  const int c = (blk >> 6) & 127;
  const int b = blk >> 13;
  const int dp = pidx >> 4, hp = (pidx >> 2) & 3, wp = pidx & 3;
  const int tid = threadIdx.x;
  const int pz = tid >> 4, py = tid & 15;
  const int base =
      (((b * C_ + c) * D_ + dp * 16 + pz) * H_ + hp * 16 + py) * W_ + wp * 16;
  const float4* pa = (const float4*)(a + base);
  const float4* pp = (const float4*)(p + base);
  float sa = 0.f, sp = 0.f;
  #pragma unroll
  for (int q = 0; q < 4; ++q) {
    float4 va = pa[q], vp = pp[q];
    sa += va.x + va.y + va.z + va.w;
    sp += vp.x + vp.y + vp.z + vp.w;
  }
  sA[tid] = sa; sB[tid] = sp;
  __syncthreads();
  for (int off = 128; off > 0; off >>= 1) {
    if (tid < off) { sA[tid] += sA[tid + off]; sB[tid] += sB[tid + off]; }
    __syncthreads();
  }
  if (tid == 0) {
    pma[blk] = sA[0] * (1.f / 4096.f);
    pmp[blk] = sB[0] * (1.f / 4096.f);
  }
}

// ---------------------------------------------------------------------------
// k3: patch-level projector + normalize + cosine loss.
// "Mixed" view: a_vec[b,n,t] = PM[b][2n + t/64][t%64].
// grid = B*NP = 128 blocks of 128 threads. Tiny -> VALU.
// ---------------------------------------------------------------------------
__global__ __launch_bounds__(128) void k3_patchproj(
    const float* __restrict__ pma, const float* __restrict__ pmp,
    const float* __restrict__ pw1, const float* __restrict__ pb1,
    const float* __restrict__ pw2, const float* __restrict__ pb2,
    float* __restrict__ patch_losses) {
  __shared__ float av[128], pv[128], ha[128], hb[128], oa[64], ob[64];
  __shared__ float scr[128], scr2[128];
  const int bn = blockIdx.x, t = threadIdx.x;
  const int b = bn >> 6, n = bn & 63;
  {
    int src = (b * C_ + (2 * n + (t >> 6))) * 64 + (t & 63);
    av[t] = pma[src];
    pv[t] = pmp[src];
  }
  __syncthreads();
  {
    const float* wr = pw1 + t * 128;
    float sa = 0.f, sp = 0.f;
    for (int k = 0; k < 128; ++k) { sa += av[k] * wr[k]; sp += pv[k] * wr[k]; }
    ha[t] = fmaxf(sa + pb1[t], 0.f);
    hb[t] = fmaxf(sp + pb1[t], 0.f);
  }
  __syncthreads();
  if (t < 64) {
    const float* wr = pw2 + t * 128;
    float sa = 0.f, sp = 0.f;
    for (int k = 0; k < 128; ++k) { sa += ha[k] * wr[k]; sp += hb[k] * wr[k]; }
    oa[t] = sa + pb2[t];
    ob[t] = sp + pb2[t];
  }
  __syncthreads();
  scr[t]  = (t < 64) ? oa[t] * oa[t] : 0.f;
  scr2[t] = (t < 64) ? ob[t] * ob[t] : 0.f;
  __syncthreads();
  for (int off = 64; off > 0; off >>= 1) {
    if (t < off) { scr[t] += scr[t + off]; scr2[t] += scr2[t + off]; }
    __syncthreads();
  }
  float inva = 1.f / fmaxf(sqrtf(scr[0]), 1e-12f);
  float invb = 1.f / fmaxf(sqrtf(scr2[0]), 1e-12f);
  __syncthreads();
  scr[t] = (t < 64) ? (oa[t] * inva) * (ob[t] * invb) : 0.f;
  __syncthreads();
  for (int off = 64; off > 0; off >>= 1) {
    if (t < off) scr[t] += scr[t + off];
    __syncthreads();
  }
  if (t == 0) patch_losses[bn] = -scr[0] * TEMP_INV;
}

// ---------------------------------------------------------------------------
// k4: voxel-level projector via WMMA.
// grid = B*NP*2*7; each block: one M-tile of 16 samples for (b,n,{a,p}).
// Weights staged pre-swizzled (16-wide c-groups land contiguous -> packed
// v8h stores); gather batched into one packed b128 store per thread.
// ---------------------------------------------------------------------------
__global__ __launch_bounds__(256) void k4_voxproj(
    const float* __restrict__ anchor, const float* __restrict__ positive,
    const float* __restrict__ pw1, const float* __restrict__ pb1,
    const float* __restrict__ pw2, const float* __restrict__ pb2,
    const int* __restrict__ voxel_idx,
    float* __restrict__ vproj) {
  __shared__ alignas(32) _Float16 w1_sw[8][4][32][16];  // [jt][kb][lane][e]
  __shared__ alignas(32) _Float16 w2_sw[4][4][32][16];
  __shared__ alignas(16) _Float16 X_lds[16][128];
  __shared__ alignas(16) _Float16 h_lds[16][128];
  __shared__ float    o_lds[16][64];
  __shared__ float    inv_lds[16];

  int t = blockIdx.x;
  const int mt = t % 7; t /= 7;
  const int av = t & 1; t >>= 1;
  const int n = t & 63;
  const int b = t >> 6;
  const float* feats = av ? positive : anchor;

  const int tid = threadIdx.x, lane = tid & 31, wave = tid >> 5;
  const int half = lane >> 4, nl = lane & 15;

  // stage w1 pre-swizzled: 16-consecutive-c groups are contiguous in the
  // fragment layout (same lane, e = 0..15)
  #pragma unroll
  for (int g = 0; g < 4; ++g) {
    int idx = tid * 64 + g * 16;       // 0..16383, 16-aligned
    int j = idx >> 7, c = idx & 127;
    const float4* src = (const float4*)(pw1 + idx);
    float4 f0 = src[0], f1 = src[1], f2 = src[2], f3 = src[3];
    _Float16* dst = &w1_sw[j >> 4][c >> 5][((c >> 4) & 1) * 16 + (j & 15)][0];
    *(v8h*)dst       = pack8(f0, f1);
    *(v8h*)(dst + 8) = pack8(f2, f3);
  }
  #pragma unroll
  for (int g = 0; g < 2; ++g) {
    int idx = tid * 32 + g * 16;       // 0..8191, 16-aligned
    int j = idx >> 7, c = idx & 127;
    const float4* src = (const float4*)(pw2 + idx);
    float4 f0 = src[0], f1 = src[1], f2 = src[2], f3 = src[3];
    _Float16* dst = &w2_sw[j >> 4][c >> 5][((c >> 4) & 1) * 16 + (j & 15)][0];
    *(v8h*)dst       = pack8(f0, f1);
    *(v8h*)(dst + 8) = pack8(f2, f3);
  }
  // gather X tile (mixed view): X[r][c'] with ch = 2n + c'/64, patch = c'%64
  {
    const int idx0 = tid * 8;
    const int r = idx0 >> 7;           // constant per thread
    const int c0 = idx0 & 127;
    const int sg = mt * 16 + r;
    float tmp[8];
    if (sg < S_) {
      int vi = voxel_idx[sg];
      int pz = (vi >> 8) & 15, py = (vi >> 4) & 15, px = vi & 15;
      #pragma unroll
      for (int q = 0; q < 8; ++q) {
        int c = c0 + q;
        int cc = 2 * n + (c >> 6);
        int pidx = c & 63;
        int zz = (pidx >> 4) * 16 + pz;
        int yy = ((pidx >> 2) & 3) * 16 + py;
        int xx = (pidx & 3) * 16 + px;
        tmp[q] = feats[(((b * C_ + cc) * D_ + zz) * H_ + yy) * W_ + xx];
      }
    } else {
      #pragma unroll
      for (int q = 0; q < 8; ++q) tmp[q] = 0.f;
    }
    v8h pk;
    #pragma unroll
    for (int q = 0; q < 8; ++q) pk[q] = (_Float16)tmp[q];
    *(v8h*)&X_lds[r][c0] = pk;
  }
  __syncthreads();
  // layer 1: h = relu(X @ w1^T + b1); each wave owns one 16-wide N-tile
  {
    v8f acc = {0.f, 0.f, 0.f, 0.f, 0.f, 0.f, 0.f, 0.f};
    const int j = wave * 16 + nl;
    #pragma unroll
    for (int kb = 0; kb < 4; ++kb) {
      v8h lo = *(const v8h*)&X_lds[nl][kb * 32 + half * 16];
      v8h hi = *(const v8h*)&X_lds[nl][kb * 32 + half * 16 + 8];
      v16h af = cat8(lo, hi);
      v16h bf = *(const v16h*)&w1_sw[wave][kb][lane][0];
      acc = __builtin_amdgcn_wmma_f32_16x16x32_f16(false, af, false, bf,
                                                   (short)0, acc, false, false);
    }
    const float bj = pb1[j];
    #pragma unroll
    for (int i = 0; i < 8; ++i)
      h_lds[half * 8 + i][j] = (_Float16)fmaxf(acc[i] + bj, 0.f);
  }
  __syncthreads();
  // layer 2: o = h @ w2^T + b2; waves 0..3 own the 64 outputs
  if (wave < 4) {
    v8f acc = {0.f, 0.f, 0.f, 0.f, 0.f, 0.f, 0.f, 0.f};
    const int j = wave * 16 + nl;
    #pragma unroll
    for (int kb = 0; kb < 4; ++kb) {
      v8h lo = *(const v8h*)&h_lds[nl][kb * 32 + half * 16];
      v8h hi = *(const v8h*)&h_lds[nl][kb * 32 + half * 16 + 8];
      v16h af = cat8(lo, hi);
      v16h bf = *(const v16h*)&w2_sw[wave][kb][lane][0];
      acc = __builtin_amdgcn_wmma_f32_16x16x32_f16(false, af, false, bf,
                                                   (short)0, acc, false, false);
    }
    const float bj = pb2[j];
    #pragma unroll
    for (int i = 0; i < 8; ++i)
      o_lds[half * 8 + i][j] = acc[i] + bj;
  }
  __syncthreads();
  if (tid < 16) {
    float ss = 0.f;
    for (int j = 0; j < 64; ++j) { float v = o_lds[tid][j]; ss += v * v; }
    inv_lds[tid] = 1.f / fmaxf(sqrtf(ss), 1e-12f);
  }
  __syncthreads();
  const int bnav = blockIdx.x / 7;     // (b*64+n)*2 + av
  #pragma unroll
  for (int q = 0; q < 4; ++q) {
    int idx = tid * 4 + q;             // 0..1023
    int r = idx >> 6, j = idx & 63;
    int sg = mt * 16 + r;
    if (sg < S_)
      vproj[((long)bnav * S_ + sg) * 64 + j] = o_lds[r][j] * inv_lds[r];
  }
}

// ---------------------------------------------------------------------------
// k5: per-(b,n) 100x100 sim, logsumexp - diag, mean -> vox_losses.
// grid = B*NP = 128 blocks of 256 threads.
// ---------------------------------------------------------------------------
__global__ __launch_bounds__(256) void k5_voxloss(
    const float* __restrict__ vproj, float* __restrict__ vox_losses) {
  __shared__ float aL[100][65];
  __shared__ float pL[100][65];
  __shared__ float scr[256];
  const int bn = blockIdx.x, t = threadIdx.x;
  const float* ab = vproj + (long)(bn * 2 + 0) * S_ * 64;
  const float* pb = vproj + (long)(bn * 2 + 1) * S_ * 64;
  for (int idx = t; idx < S_ * 64; idx += 256) {
    int s = idx >> 6, j = idx & 63;
    aL[s][j] = ab[idx];
    pL[s][j] = pb[idx];
  }
  __syncthreads();
  float acc = 0.f;
  for (int i = 0; i < S_; ++i) {
    float s = -1e30f;
    if (t < S_) {
      float d = 0.f;
      #pragma unroll 8
      for (int k = 0; k < 64; ++k) d += aL[i][k] * pL[t][k];
      s = d * TEMP_INV;
    }
    scr[t] = s;
    __syncthreads();
    for (int off = 128; off > 0; off >>= 1) {
      if (t < off) scr[t] = fmaxf(scr[t], scr[t + off]);
      __syncthreads();
    }
    float m = scr[0];
    __syncthreads();
    scr[t] = (t < S_) ? expf(s - m) : 0.f;
    __syncthreads();
    for (int off = 128; off > 0; off >>= 1) {
      if (t < off) scr[t] += scr[t + off];
      __syncthreads();
    }
    float lse = m + logf(scr[0]);
    __syncthreads();
    if (t == i) acc += lse - s;
  }
  scr[t] = acc;
  __syncthreads();
  for (int off = 128; off > 0; off >>= 1) {
    if (t < off) scr[t] += scr[t + off];
    __syncthreads();
  }
  if (t == 0) vox_losses[bn] = scr[0] * (1.f / S_);
}

// ---------------------------------------------------------------------------
// k6: masked weighted reduction -> scalar output.
// ---------------------------------------------------------------------------
__global__ __launch_bounds__(128) void k6_final(
    const int* __restrict__ edge_counts,
    const float* __restrict__ patch_losses,
    const float* __restrict__ vox_losses,
    const float* __restrict__ lw,
    float* __restrict__ out) {
  __shared__ float sp[128], sv[128], sc[128];
  const int t = threadIdx.x;
  float ratio = edge_counts[t] * (1.f / 4096.f);
  float mvox = (ratio > 0.6f) ? 1.f : 0.f;
  float mpat = (ratio < 0.4f) ? 1.f : 0.f;
  sp[t] = patch_losses[t] * mpat;
  sv[t] = vox_losses[t] * mvox;
  sc[t] = mpat + mvox;
  __syncthreads();
  for (int off = 64; off > 0; off >>= 1) {
    if (t < off) { sp[t] += sp[t + off]; sv[t] += sv[t + off]; sc[t] += sc[t + off]; }
    __syncthreads();
  }
  if (t == 0) {
    float total = lw[0] * sp[0] + lw[1] * sv[0];
    out[0] = (sc[0] > 0.f) ? total / fmaxf(sc[0], 1.f) : 0.f;
  }
}

// ---------------------------------------------------------------------------
extern "C" void kernel_launch(void* const* d_in, const int* in_sizes, int n_in,
                              void* d_out, int out_size, void* d_ws, size_t ws_size,
                              hipStream_t stream) {
  (void)in_sizes; (void)n_in; (void)out_size; (void)ws_size;
  const float* anchor   = (const float*)d_in[0];
  const float* positive = (const float*)d_in[1];
  const float* cw1 = (const float*)d_in[2];
  const float* cb1 = (const float*)d_in[3];
  const float* cw2 = (const float*)d_in[4];
  const float* cb2 = (const float*)d_in[5];
  const float* pw1 = (const float*)d_in[6];
  const float* pb1 = (const float*)d_in[7];
  const float* pw2 = (const float*)d_in[8];
  const float* pb2 = (const float*)d_in[9];
  const float* lw  = (const float*)d_in[10];
  const int* voxel_idx = (const int*)d_in[11];
  float* out = (float*)d_out;

  // workspace layout
  char* ws = (char*)d_ws;
  int*   edge_counts  = (int*)ws;                          // 128 * 4B
  float* pma          = (float*)(ws + 512);                // 16384 floats
  float* pmp          = (float*)(ws + 512 + 65536);        // 16384 floats
  float* patch_losses = (float*)(ws + 512 + 131072);       // 128 floats
  float* vox_losses   = (float*)(ws + 512 + 131072 + 512); // 128 floats
  float* vproj        = (float*)(ws + 512 + 131072 + 1024);// 128*2*100*64 floats

  k0_zero<<<1, 128, 0, stream>>>(edge_counts);
  k1_region<<<B_ * D_ * H_, 256, 0, stream>>>(anchor, cw1, cb1, cw2, cb2, edge_counts);
  k2_means<<<B_ * C_ * 64, 256, 0, stream>>>(anchor, positive, pma, pmp);
  k3_patchproj<<<B_ * NP, 128, 0, stream>>>(pma, pmp, pw1, pb1, pw2, pb2, patch_losses);
  k4_voxproj<<<B_ * NP * 2 * 7, 256, 0, stream>>>(anchor, positive, pw1, pb1, pw2, pb2,
                                                  voxel_idx, vproj);
  k5_voxloss<<<B_ * NP, 256, 0, stream>>>(vproj, vox_losses);
  k6_final<<<1, 128, 0, stream>>>(edge_counts, patch_losses, vox_losses, lw, out);
}
